// SparseMMDiTBlock_7567732376029
// MI455X (gfx1250) — matrix operand order repaired and checked by reading, not compiled, not run on hardware
//
#include <hip/hip_runtime.h>
#include <math.h>

// ---------------- dims ----------------
#define S_LEN 5120
#define T_LEN 256
#define DIMC  1152
#define HEADS 16
#define HD    72
#define HDP   96          // padded head dim (mult of 32 for WMMA K, of 16 for N)
#define G_LEN 1280        // S/4
#define L_TOT 1536        // G_LEN + T_LEN
#define BHN   64          // 4 * HEADS
#define INNERC 4608
#define MODC  6912        // 6*DIM

typedef __bf16 bf16;
typedef __attribute__((ext_vector_type(16))) __bf16 v16bf;
typedef __attribute__((ext_vector_type(8)))  __bf16 v8bf;
typedef __attribute__((ext_vector_type(8)))  float  v8f;

union Frag { v16bf v; v8bf h[2]; };

__device__ __forceinline__ bf16 f2bf(float f) {
  union { float f; unsigned u; } in; in.f = f;
  unsigned lsb = (in.u >> 16) & 1u;
  unsigned r = in.u + 0x7fffu + lsb;              // round-to-nearest-even
  union { unsigned short u; bf16 b; } out; out.u = (unsigned short)(r >> 16);
  return out.b;
}

__device__ __forceinline__ v8f wmma_bf(v16bf a, v16bf b, v8f c) {
  return __builtin_amdgcn_wmma_f32_16x16x32_bf16(false, a, false, b, (short)0, c, false, false);
}

// ---- DPP16 butterfly reductions within 16-lane groups (no LDS, no waits) ----
// quad_perm [1,0,3,2] = 0xB1 (xor1), quad_perm [2,3,0,1] = 0x4E (xor2),
// row_half_mirror = 0x141 (xor4 once quads uniform), row_mirror = 0x140 (xor8).
template<int CTRL>
__device__ __forceinline__ float dpp_f32(float x) {
  int i = __builtin_bit_cast(int, x);
  int p = __builtin_amdgcn_update_dpp(0, i, CTRL, 0xf, 0xf, true);
  return __builtin_bit_cast(float, p);
}
__device__ __forceinline__ float red_max16(float v) {
  v = fmaxf(v, dpp_f32<0xB1>(v));
  v = fmaxf(v, dpp_f32<0x4E>(v));
  v = fmaxf(v, dpp_f32<0x141>(v));
  v = fmaxf(v, dpp_f32<0x140>(v));
  return v;
}
__device__ __forceinline__ float red_sum16(float v) {
  v += dpp_f32<0xB1>(v);
  v += dpp_f32<0x4E>(v);
  v += dpp_f32<0x141>(v);
  v += dpp_f32<0x140>(v);
  return v;
}

// ---------------- adaLN modulation: t = silu(temb) @ lw + lb ----------------
__global__ void mod_kernel(const float* __restrict__ temb, const float* __restrict__ lw,
                           const float* __restrict__ lb, float* __restrict__ t) {
  __shared__ float st[DIMC];
  for (int i = threadIdx.x; i < DIMC; i += 256) {
    float x = temb[i];
    st[i] = x / (1.f + __expf(-x));
  }
  __syncthreads();
  int c = blockIdx.x * 256 + threadIdx.x;          // < 6912
  float acc = lb[c];
  for (int k = 0; k < DIMC; ++k)
    acc = fmaf(st[k], lw[(size_t)k * MODC + c], acc);
  t[c] = acc;
}

// ---------------- LayerNorm + scale/shift, fp32 -> bf16 ----------------
__global__ void ln_mod_kernel(const float* __restrict__ X, const float* __restrict__ w,
                              const float* __restrict__ b, const float* __restrict__ sh,
                              const float* __restrict__ sc, bf16* __restrict__ Y) {
  int row = blockIdx.x, tid = threadIdx.x;         // 288 threads, 4 elems each
  const float* x = X + (size_t)row * DIMC;
  float v[4]; float s1 = 0.f, s2 = 0.f;
#pragma unroll
  for (int j = 0; j < 4; ++j) { v[j] = x[tid + j * 288]; s1 += v[j]; s2 += v[j] * v[j]; }
  for (int off = 16; off > 0; off >>= 1) { s1 += __shfl_xor(s1, off); s2 += __shfl_xor(s2, off); }
  __shared__ float ws1[9], ws2[9];
  __shared__ float mu, inv;
  int wid = tid >> 5, lane = tid & 31;
  if (lane == 0) { ws1[wid] = s1; ws2[wid] = s2; }
  __syncthreads();
  if (tid == 0) {
    float a = 0.f, c2 = 0.f;
    for (int i = 0; i < 9; ++i) { a += ws1[i]; c2 += ws2[i]; }
    float m = a / (float)DIMC;
    mu = m; inv = rsqrtf(c2 / (float)DIMC - m * m + 1e-5f);
  }
  __syncthreads();
#pragma unroll
  for (int j = 0; j < 4; ++j) {
    int col = tid + j * 288;
    float n = (v[j] - mu) * inv * w[col] + b[col];
    n = n * (1.f + sc[col]) + sh[col];
    Y[(size_t)row * DIMC + col] = f2bf(n);
  }
}

// ---------------- weight convert + transpose: Wt[n][k] = bf16(W[k][n]) ----------------
__global__ void transpose_cvt(const float* __restrict__ W, bf16* __restrict__ Wt, int K, int N) {
  __shared__ float tile[32][33];
  int k0 = blockIdx.y * 32, n0 = blockIdx.x * 32;
  int tx = threadIdx.x & 31, ty = threadIdx.x >> 5;   // 256 thr: 32x8
  for (int r = ty; r < 32; r += 8) tile[r][tx] = W[(size_t)(k0 + r) * N + n0 + tx];
  __syncthreads();
  for (int r = ty; r < 32; r += 8) Wt[(size_t)(n0 + r) * K + k0 + tx] = f2bf(tile[tx][r]);
}

// ---------------- GEMM: C = A[MxK] @ Bt^T, bf16 WMMA, fp32 acc ----------------
// Wave tile: 32 rows x 64 cols (2 A-frags share each B-frag: 8 WMMAs / 12 loads).
// MODE 0: oA[row][col] = acc
// MODE 1: out = res + gate[col]*(acc + bias[col]), row-split at Msplit (A/B sets)
template<int MODE>
__global__ void gemm_bf16(const bf16* __restrict__ A, const bf16* __restrict__ Bt,
                          int K, int N, const float* __restrict__ bias,
                          const float* __restrict__ gA, const float* __restrict__ gB,
                          const float* __restrict__ rA, const float* __restrict__ rB,
                          float* __restrict__ oA, float* __restrict__ oB, int Msplit) {
  int wid = threadIdx.x >> 5, lane = threadIdx.x & 31;
  int r = lane & 15, hi = lane >> 4;
  int row0 = blockIdx.y * 256 + wid * 32;
  int col0 = blockIdx.x * 64;
  v8f acc[2][4];
#pragma unroll
  for (int mt = 0; mt < 2; ++mt)
    for (int t = 0; t < 4; ++t)
      for (int i = 0; i < 8; ++i) acc[mt][t][i] = 0.f;
  const bf16* arow0 = A + (size_t)(row0 + r) * K;
  const bf16* arow1 = A + (size_t)(row0 + 16 + r) * K;
  for (int k0 = 0; k0 < K; k0 += 32) {
    Frag a0, a1;
    a0.h[0] = *reinterpret_cast<const v8bf*>(arow0 + k0 + hi * 8);
    a0.h[1] = *reinterpret_cast<const v8bf*>(arow0 + k0 + 16 + hi * 8);
    a1.h[0] = *reinterpret_cast<const v8bf*>(arow1 + k0 + hi * 8);
    a1.h[1] = *reinterpret_cast<const v8bf*>(arow1 + k0 + 16 + hi * 8);
#pragma unroll
    for (int nt = 0; nt < 4; ++nt) {
      const bf16* brow = Bt + (size_t)(col0 + nt * 16 + r) * K + k0 + hi * 8;
      Frag b;
      b.h[0] = *reinterpret_cast<const v8bf*>(brow);
      b.h[1] = *reinterpret_cast<const v8bf*>(brow + 16);
      acc[0][nt] = wmma_bf(a0.v, b.v, acc[0][nt]);
      acc[1][nt] = wmma_bf(a1.v, b.v, acc[1][nt]);
    }
  }
#pragma unroll
  for (int mt = 0; mt < 2; ++mt) {
#pragma unroll
    for (int nt = 0; nt < 4; ++nt) {
#pragma unroll
      for (int i = 0; i < 8; ++i) {
        int row = row0 + mt * 16 + i + hi * 8;
        int col = col0 + nt * 16 + r;
        float v = acc[mt][nt][i];
        if (MODE == 0) {
          oA[(size_t)row * N + col] = v;
        } else {
          if (bias) v += bias[col];
          if (row < Msplit) {
            oA[(size_t)row * N + col] = rA[(size_t)row * N + col] + gA[col] * v;
          } else {
            int rr = row - Msplit;
            oB[(size_t)rr * N + col] = rB[(size_t)rr * N + col] + gB[col] * v;
          }
        }
      }
    }
  }
}

// ---------------- QKV post-process: RMS + RoPE + pack to padded bf16 layouts ----------------
__global__ void qkv_prep(const float* __restrict__ qh, const float* __restrict__ kh,
                         const float* __restrict__ vh, const float* __restrict__ aqh,
                         const float* __restrict__ akh, const float* __restrict__ avh,
                         const float* __restrict__ cosb, const float* __restrict__ sinb,
                         const float* __restrict__ nqw, const float* __restrict__ nkw,
                         const float* __restrict__ naqw, const float* __restrict__ nakw,
                         bf16* __restrict__ qf, bf16* __restrict__ kf, bf16* __restrict__ vt) {
  int s = blockIdx.x, h = blockIdx.y, d = threadIdx.x;   // d: 0..95
  bool isH = s < S_LEN;
  const float *qs, *ks, *vs, *wq, *wk;
  if (isH) {
    size_t o = (size_t)s * DIMC + h * HD;
    qs = qh + o; ks = kh + o; vs = vh + o; wq = nqw; wk = nkw;
  } else {
    size_t o = (size_t)(s - S_LEN) * DIMC + h * HD;
    qs = aqh + o; ks = akh + o; vs = avh + o; wq = naqw; wk = nakw;
  }
  float qv = (d < HD) ? qs[d] : 0.f;
  float kv = (d < HD) ? ks[d] : 0.f;
  float vv = (d < HD) ? vs[d] : 0.f;
  __shared__ float sq[96], sk[96];
  sq[d] = qv * qv; sk[d] = kv * kv;
  __syncthreads();
  __shared__ float rq, rk;
  if (d == 0) {
    float a = 0.f, b = 0.f;
    for (int i = 0; i < HD; ++i) { a += sq[i]; b += sk[i]; }
    rq = rsqrtf(a / (float)HD + 1e-5f);
    rk = rsqrtf(b / (float)HD + 1e-5f);
  }
  __syncthreads();
  float qn = 0.f, kn = 0.f;
  if (d < HD) { qn = qv * rq * wq[d]; kn = kv * rk * wk[d]; }
  if (isH && d < HD) {
    float c = cosb[(size_t)s * HD + d], sn = sinb[(size_t)s * HD + d];
    int pd = (d < 36) ? d + 36 : d - 36;
    float qp = qs[pd] * rq * wq[pd];
    float kp = ks[pd] * rk * wk[pd];
    float qr = (d < 36) ? -qp : qp;
    float kr = (d < 36) ? -kp : kp;
    qn = qn * c + qr * sn;
    kn = kn * c + kr * sn;
  }
  const float SC = 0.11785113019775792f;   // 1/sqrt(72), folded into Q
  if (isH) {
    int l = s >> 2, bp = s & 3, bh = bp * HEADS + h;
    size_t base = ((size_t)bh * L_TOT + l) * HDP + d;
    qf[base] = f2bf(qn * SC);
    kf[base] = f2bf(kn);
    vt[((size_t)bh * HDP + d) * L_TOT + l] = f2bf(vv);
  } else {
    int l = G_LEN + (s - S_LEN);
#pragma unroll
    for (int bp = 0; bp < 4; ++bp) {
      int bh = bp * HEADS + h;
      size_t base = ((size_t)bh * L_TOT + l) * HDP + d;
      qf[base] = f2bf(qn * SC);
      kf[base] = f2bf(kn);
      vt[((size_t)bh * HDP + d) * L_TOT + l] = f2bf(vv);
    }
  }
}

// ---------------- Flash attention: per (bh, 16-query tile), online softmax ----------------
__global__ void __launch_bounds__(128, 1)
flash_attn(const bf16* __restrict__ qf, const bf16* __restrict__ kf,
           const bf16* __restrict__ vt, bf16* __restrict__ ovbf,
           float* __restrict__ oef) {
  int wid = threadIdx.x >> 5, lane = threadIdx.x & 31;
  int r = lane & 15, hi = lane >> 4;
  int qt = blockIdx.x * 4 + wid;     // 0..95
  int bh = blockIdx.y;               // 0..63
  const bf16* Q  = qf + (size_t)bh * L_TOT * HDP;
  const bf16* Kb = kf + (size_t)bh * L_TOT * HDP;
  const bf16* Vt = vt + (size_t)bh * HDP * L_TOT;

  Frag qa[3];
  {
    size_t qro = (size_t)(qt * 16 + r) * HDP;
#pragma unroll
    for (int t = 0; t < 3; ++t) {
      qa[t].h[0] = *reinterpret_cast<const v8bf*>(Q + qro + t * 32 + hi * 16);
      qa[t].h[1] = *reinterpret_cast<const v8bf*>(Q + qro + t * 32 + hi * 16 + 8);
    }
  }
  v8f o[6];
#pragma unroll
  for (int nt = 0; nt < 6; ++nt) for (int i = 0; i < 8; ++i) o[nt][i] = 0.f;
  float m_i[8], l_i[8];
#pragma unroll
  for (int i = 0; i < 8; ++i) { m_i[i] = -1e30f; l_i[i] = 0.f; }

  __shared__ bf16 Plds[4][16 * 32];
  bf16* P = Plds[wid];

  for (int kt = 0; kt < L_TOT / 32; ++kt) {
    int key0 = kt * 32;
    v8f s0, s1;
#pragma unroll
    for (int i = 0; i < 8; ++i) { s0[i] = 0.f; s1[i] = 0.f; }
#pragma unroll
    for (int t = 0; t < 3; ++t) {
      int d0 = t * 32;
      Frag b0, b1;
      const bf16* k0p = Kb + (size_t)(key0 + r) * HDP + d0 + hi * 16;
      const bf16* k1p = Kb + (size_t)(key0 + 16 + r) * HDP + d0 + hi * 16;
      b0.h[0] = *reinterpret_cast<const v8bf*>(k0p);
      b0.h[1] = *reinterpret_cast<const v8bf*>(k0p + 8);
      b1.h[0] = *reinterpret_cast<const v8bf*>(k1p);
      b1.h[1] = *reinterpret_cast<const v8bf*>(k1p + 8);
      s0 = wmma_bf(qa[t].v, b0.v, s0);
      s1 = wmma_bf(qa[t].v, b1.v, s1);
    }
    float p0[8], p1[8];
#pragma unroll
    for (int i = 0; i < 8; ++i) {
      float mx = red_max16(fmaxf(s0[i], s1[i]));
      float mnew = fmaxf(m_i[i], mx);
      float corr = __expf(m_i[i] - mnew);
      p0[i] = __expf(s0[i] - mnew);
      p1[i] = __expf(s1[i] - mnew);
      float rs = red_sum16(p0[i] + p1[i]);
      l_i[i] = l_i[i] * corr + rs;
      m_i[i] = mnew;
#pragma unroll
      for (int nt = 0; nt < 6; ++nt) o[nt][i] *= corr;
    }
    // re-layout P (C-frag -> A-frag) through LDS
#pragma unroll
    for (int i = 0; i < 8; ++i) {
      P[(i + hi * 8) * 32 + r]      = f2bf(p0[i]);
      P[(i + hi * 8) * 32 + 16 + r] = f2bf(p1[i]);
    }
    __syncthreads();
    Frag pa;
    pa.h[0] = *reinterpret_cast<const v8bf*>(P + r * 32 + hi * 16);
    pa.h[1] = *reinterpret_cast<const v8bf*>(P + r * 32 + hi * 16 + 8);
    __syncthreads();
#pragma unroll
    for (int nt = 0; nt < 6; ++nt) {
      const bf16* vp = Vt + (size_t)(nt * 16 + r) * L_TOT + key0 + hi * 16;
      Frag vb;
      vb.h[0] = *reinterpret_cast<const v8bf*>(vp);
      vb.h[1] = *reinterpret_cast<const v8bf*>(vp + 8);
      o[nt] = wmma_bf(pa.v, vb.v, o[nt]);
    }
  }
  // epilogue: divide by l, scatter (drop pad)
  int bp = bh >> 4, h = bh & 15;
#pragma unroll
  for (int i = 0; i < 8; ++i) {
    float invl = 1.f / l_i[i];
    int lrow = qt * 16 + i + hi * 8;
#pragma unroll
    for (int nt = 0; nt < 6; ++nt) {
      int d = nt * 16 + r;
      if (d < HD) {
        float val = o[nt][i] * invl;
        if (lrow < G_LEN)
          ovbf[(size_t)(lrow * 4 + bp) * DIMC + h * HD + d] = f2bf(val);
        else
          oef[((size_t)(lrow - G_LEN) * 4 + bp) * DIMC + h * HD + d] = val;
      }
    }
  }
}

// ---------------- encoder output mean over SPARSE_N ----------------
__global__ void oe_mean(const float* __restrict__ oef, bf16* __restrict__ oebf) {
  int idx = blockIdx.x * 256 + threadIdx.x;   // < 256*1152
  int t = idx / DIMC, c = idx % DIMC;
  size_t b = (size_t)t * 4 * DIMC + c;
  float a = 0.25f * (oef[b] + oef[b + DIMC] + oef[b + 2 * DIMC] + oef[b + 3 * DIMC]);
  oebf[idx] = f2bf(a);
}

// ---------------- FF1 fused GEGLU: P = (X@W1h) * gelu(X@W1g), bf16 out ----------------
// Wave tile: 32 rows x 32 cols, dual (h, gate) accumulators.
__global__ void ff1_geglu(const bf16* __restrict__ A, const bf16* __restrict__ W1T,
                          bf16* __restrict__ P) {
  const int K = DIMC;
  int wid = threadIdx.x >> 5, lane = threadIdx.x & 31;
  int r = lane & 15, hi = lane >> 4;
  int row0 = blockIdx.y * 256 + wid * 32;
  int col0 = blockIdx.x * 32;
  v8f ah[2][2], ag[2][2];
#pragma unroll
  for (int mt = 0; mt < 2; ++mt)
    for (int t = 0; t < 2; ++t)
      for (int i = 0; i < 8; ++i) { ah[mt][t][i] = 0.f; ag[mt][t][i] = 0.f; }
  const bf16* arow0 = A + (size_t)(row0 + r) * K;
  const bf16* arow1 = A + (size_t)(row0 + 16 + r) * K;
  for (int k0 = 0; k0 < K; k0 += 32) {
    Frag a0, a1;
    a0.h[0] = *reinterpret_cast<const v8bf*>(arow0 + k0 + hi * 16);
    a0.h[1] = *reinterpret_cast<const v8bf*>(arow0 + k0 + hi * 16 + 8);
    a1.h[0] = *reinterpret_cast<const v8bf*>(arow1 + k0 + hi * 16);
    a1.h[1] = *reinterpret_cast<const v8bf*>(arow1 + k0 + hi * 16 + 8);
#pragma unroll
    for (int nt = 0; nt < 2; ++nt) {
      int cn = col0 + nt * 16 + r;
      const bf16* bh_p = W1T + (size_t)cn * K + k0 + hi * 16;
      const bf16* bg_p = W1T + (size_t)(INNERC + cn) * K + k0 + hi * 16;
      Frag bh_, bg_;
      bh_.h[0] = *reinterpret_cast<const v8bf*>(bh_p);
      bh_.h[1] = *reinterpret_cast<const v8bf*>(bh_p + 8);
      bg_.h[0] = *reinterpret_cast<const v8bf*>(bg_p);
      bg_.h[1] = *reinterpret_cast<const v8bf*>(bg_p + 8);
      ah[0][nt] = wmma_bf(a0.v, bh_.v, ah[0][nt]);
      ah[1][nt] = wmma_bf(a1.v, bh_.v, ah[1][nt]);
      ag[0][nt] = wmma_bf(a0.v, bg_.v, ag[0][nt]);
      ag[1][nt] = wmma_bf(a1.v, bg_.v, ag[1][nt]);
    }
  }
#pragma unroll
  for (int mt = 0; mt < 2; ++mt) {
#pragma unroll
    for (int nt = 0; nt < 2; ++nt) {
#pragma unroll
      for (int i = 0; i < 8; ++i) {
        int row = row0 + mt * 16 + i + hi * 8;
        int col = col0 + nt * 16 + r;
        float g = ag[mt][nt][i];
        float gel = 0.5f * g * (1.f + erff(g * 0.70710678118654752f));
        P[(size_t)row * INNERC + col] = f2bf(ah[mt][nt][i] * gel);
      }
    }
  }
}

// ==================================================================
extern "C" void kernel_launch(void* const* d_in, const int* in_sizes, int n_in,
                              void* d_out, int out_size, void* d_ws, size_t ws_size,
                              hipStream_t stream) {
  const float* h0    = (const float*)d_in[0];
  const float* e0    = (const float*)d_in[1];
  const float* temb  = (const float*)d_in[2];
  const float* cosb  = (const float*)d_in[3];
  const float* sinb  = (const float*)d_in[4];
  const float* n1lw  = (const float*)d_in[5];
  const float* n1lb  = (const float*)d_in[6];
  const float* n1w   = (const float*)d_in[7];
  const float* n1b   = (const float*)d_in[8];
  const float* n1ew  = (const float*)d_in[9];
  const float* n1eb  = (const float*)d_in[10];
  const float* qw    = (const float*)d_in[11];
  const float* kw    = (const float*)d_in[12];
  const float* vw    = (const float*)d_in[13];
  const float* aqw   = (const float*)d_in[14];
  const float* akw   = (const float*)d_in[15];
  const float* avw   = (const float*)d_in[16];
  const float* nqw   = (const float*)d_in[17];
  const float* nkw   = (const float*)d_in[18];
  const float* naqw  = (const float*)d_in[19];
  const float* nakw  = (const float*)d_in[20];
  const float* outw  = (const float*)d_in[21];
  const float* outb  = (const float*)d_in[22];
  const float* aoutw = (const float*)d_in[23];
  const float* aoutb = (const float*)d_in[24];
  const float* n2lw  = (const float*)d_in[25];
  const float* n2lb  = (const float*)d_in[26];
  const float* n2w   = (const float*)d_in[27];
  const float* n2b   = (const float*)d_in[28];
  const float* n2ew  = (const float*)d_in[29];
  const float* n2eb  = (const float*)d_in[30];
  const float* ffw1  = (const float*)d_in[31];
  const float* ffw2  = (const float*)d_in[32];

  char* ws = (char*)d_ws;
  size_t off = 0;
  auto alloc = [&](size_t bytes) -> void* {
    void* p = ws + off;
    off += (bytes + 255) & ~(size_t)255;
    return p;
  };

  float* t1  = (float*)alloc((size_t)MODC * 4);
  float* t2  = (float*)alloc((size_t)MODC * 4);
  bf16* Xbf  = (bf16*)alloc((size_t)S_LEN * DIMC * 2);
  bf16* Ebf  = (bf16*)alloc((size_t)T_LEN * DIMC * 2);
  const size_t WSZ = (size_t)DIMC * DIMC * 2;
  bf16* WqT  = (bf16*)alloc(WSZ);
  bf16* WkT  = (bf16*)alloc(WSZ);
  bf16* WvT  = (bf16*)alloc(WSZ);
  bf16* WaqT = (bf16*)alloc(WSZ);
  bf16* WakT = (bf16*)alloc(WSZ);
  bf16* WavT = (bf16*)alloc(WSZ);
  bf16* WoT  = (bf16*)alloc(WSZ);
  bf16* WaoT = (bf16*)alloc(WSZ);
  bf16* W1T  = (bf16*)alloc((size_t)(2 * INNERC) * DIMC * 2);
  bf16* W2T  = (bf16*)alloc((size_t)DIMC * INNERC * 2);
  size_t regionA = off;                              // qh/kh/vh -> later Pbf
  float* qh  = (float*)alloc((size_t)S_LEN * DIMC * 4);
  float* kh  = (float*)alloc((size_t)S_LEN * DIMC * 4);
  float* vh  = (float*)alloc((size_t)S_LEN * DIMC * 4);
  float* aqh = (float*)alloc((size_t)T_LEN * DIMC * 4);
  float* akh = (float*)alloc((size_t)T_LEN * DIMC * 4);
  float* avh = (float*)alloc((size_t)T_LEN * DIMC * 4);
  size_t regionB = off;                              // qf/kf/vt -> later X2bf
  bf16* qf   = (bf16*)alloc((size_t)BHN * L_TOT * HDP * 2);
  bf16* kf   = (bf16*)alloc((size_t)BHN * L_TOT * HDP * 2);
  bf16* vt   = (bf16*)alloc((size_t)BHN * HDP * L_TOT * 2);
  bf16* ovbf = (bf16*)alloc((size_t)S_LEN * DIMC * 2);
  float* oef = (float*)alloc((size_t)T_LEN * 4 * DIMC * 4);
  bf16* oebf = (bf16*)alloc((size_t)T_LEN * DIMC * 2);
  float* h1  = (float*)alloc((size_t)S_LEN * DIMC * 4);
  float* e1  = (float*)alloc((size_t)T_LEN * DIMC * 4);
  bf16* Pbf  = (bf16*)(ws + regionA);                // 5376x4608 bf16, aliases q/k/vh
  bf16* X2bf = (bf16*)(ws + regionB);                // 5376x1152 bf16, aliases qf

  float* out_h = (float*)d_out;
  float* out_e = out_h + (size_t)S_LEN * DIMC;

  // 1) modulation vectors
  mod_kernel<<<dim3(MODC / 256), 256, 0, stream>>>(temb, n1lw, n1lb, t1);
  mod_kernel<<<dim3(MODC / 256), 256, 0, stream>>>(temb, n2lw, n2lb, t2);

  // 2) weight convert+transpose to bf16 N-major
  auto T = [&](const float* W, bf16* Wt, int K, int N) {
    transpose_cvt<<<dim3(N / 32, K / 32), 256, 0, stream>>>(W, Wt, K, N);
  };
  T(qw, WqT, DIMC, DIMC);   T(kw, WkT, DIMC, DIMC);   T(vw, WvT, DIMC, DIMC);
  T(aqw, WaqT, DIMC, DIMC); T(akw, WakT, DIMC, DIMC); T(avw, WavT, DIMC, DIMC);
  T(outw, WoT, DIMC, DIMC); T(aoutw, WaoT, DIMC, DIMC);
  T(ffw1, W1T, DIMC, 2 * INNERC);
  T(ffw2, W2T, INNERC, DIMC);

  // 3) norm1 + modulation
  ln_mod_kernel<<<S_LEN, 288, 0, stream>>>(h0, n1w, n1b, t1, t1 + DIMC, Xbf);
  ln_mod_kernel<<<T_LEN, 288, 0, stream>>>(e0, n1ew, n1eb, t1 + 3 * DIMC, t1 + 4 * DIMC, Ebf);

  // 4) QKV projections (bf16 WMMA, fp32 out), 256-row blocks
  gemm_bf16<0><<<dim3(DIMC / 64, S_LEN / 256), 256, 0, stream>>>(Xbf, WqT, DIMC, DIMC, nullptr, nullptr, nullptr, nullptr, nullptr, qh, qh, S_LEN);
  gemm_bf16<0><<<dim3(DIMC / 64, S_LEN / 256), 256, 0, stream>>>(Xbf, WkT, DIMC, DIMC, nullptr, nullptr, nullptr, nullptr, nullptr, kh, kh, S_LEN);
  gemm_bf16<0><<<dim3(DIMC / 64, S_LEN / 256), 256, 0, stream>>>(Xbf, WvT, DIMC, DIMC, nullptr, nullptr, nullptr, nullptr, nullptr, vh, vh, S_LEN);
  gemm_bf16<0><<<dim3(DIMC / 64, T_LEN / 256), 256, 0, stream>>>(Ebf, WaqT, DIMC, DIMC, nullptr, nullptr, nullptr, nullptr, nullptr, aqh, aqh, T_LEN);
  gemm_bf16<0><<<dim3(DIMC / 64, T_LEN / 256), 256, 0, stream>>>(Ebf, WakT, DIMC, DIMC, nullptr, nullptr, nullptr, nullptr, nullptr, akh, akh, T_LEN);
  gemm_bf16<0><<<dim3(DIMC / 64, T_LEN / 256), 256, 0, stream>>>(Ebf, WavT, DIMC, DIMC, nullptr, nullptr, nullptr, nullptr, nullptr, avh, avh, T_LEN);

  // 5) RMS + RoPE + sparse grouping into padded bf16 layouts
  qkv_prep<<<dim3(S_LEN + T_LEN, HEADS), 96, 0, stream>>>(qh, kh, vh, aqh, akh, avh,
      cosb, sinb, nqw, nkw, naqw, nakw, qf, kf, vt);

  // 6) flash attention over 64 head-matrices of 1536x1536
  flash_attn<<<dim3((L_TOT / 16) / 4, BHN), 128, 0, stream>>>(qf, kf, vt, ovbf, oef);

  // 7) encoder mean over SPARSE_N
  oe_mean<<<dim3((T_LEN * DIMC) / 256), 256, 0, stream>>>(oef, oebf);

  // 8) output projections with gated residual (h1 = h0 + gm*(ov@Wo+b))
  gemm_bf16<1><<<dim3(DIMC / 64, S_LEN / 256), 256, 0, stream>>>(ovbf, WoT, DIMC, DIMC,
      outb, t1 + 2 * DIMC, t1 + 2 * DIMC, h0, h0, h1, h1, S_LEN);
  gemm_bf16<1><<<dim3(DIMC / 64, T_LEN / 256), 256, 0, stream>>>(oebf, WaoT, DIMC, DIMC,
      aoutb, t1 + 5 * DIMC, t1 + 5 * DIMC, e0, e0, e1, e1, T_LEN);

  // 9) norm2 + modulation, concat [nh2; ne2]
  ln_mod_kernel<<<S_LEN, 288, 0, stream>>>(h1, n2w, n2b, t2, t2 + DIMC, X2bf);
  ln_mod_kernel<<<T_LEN, 288, 0, stream>>>(e1, n2ew, n2eb, t2 + 3 * DIMC, t2 + 4 * DIMC,
                                           X2bf + (size_t)S_LEN * DIMC);

  // 10) fused GEGLU FF1 (32x32 wave tile, dual accumulators)
  ff1_geglu<<<dim3(INNERC / 32, (S_LEN + T_LEN) / 256), 256, 0, stream>>>(X2bf, W1T, Pbf);

  // 11) FF2 + gated residual, split rows into h/e outputs (final result)
  gemm_bf16<1><<<dim3(DIMC / 64, (S_LEN + T_LEN) / 256), 256, 0, stream>>>(Pbf, W2T,
      INNERC, DIMC, nullptr, t2 + 2 * DIMC, t2 + 5 * DIMC, h1, e1, out_h, out_e, S_LEN);
}
